// BiLSTM_CRF_75874892251427
// MI455X (gfx1250) — compile-verified, matrix operand order checked by reading, not run on hardware
//
#include <hip/hip_runtime.h>

#define B_    32
#define T_    512
#define E_    256
#define H_    256
#define G4_   1024
#define K_    24
#define KP_   32
#define START_ 22
#define STOP_  23
#define NEGV  (-10000.0f)

typedef __attribute__((ext_vector_type(16))) __bf16 v16bf;
typedef __attribute__((ext_vector_type(8)))  __bf16 v8bf;
typedef __attribute__((ext_vector_type(8)))  float  v8f;

__device__ __forceinline__ __bf16 f2bf(float f) {
    union { float f; unsigned u; } in; in.f = f;
    unsigned r = in.u + 0x7FFFu + ((in.u >> 16) & 1u);   // round-to-nearest-even
    union { unsigned short s; __bf16 b; } out; out.s = (unsigned short)(r >> 16);
    return out.b;
}

__device__ __forceinline__ float sigm(float x) { return 1.0f / (1.0f + __expf(-x)); }
// tanh via one v_exp_f32; saturates to +/-1 correctly for |x| large.
__device__ __forceinline__ float tanh_fast(float x) {
    float e = __expf(2.0f * x);
    return 1.0f - 2.0f / (e + 1.0f);
}

union FragU { v16bf v; v8bf h[2]; };

// ISA 7.12.2 16-bit A/B layout: lane holds row (or col) = lane%16; elements 0..7
// are K = half*8 + 0..7, elements 8..15 are K = 16 + half*8 + 0..7, half = lane/16.
// Caller passes p = base + kbase + half*8 (16B aligned).
__device__ __forceinline__ v16bf load_frag(const __bf16* p) {
    FragU f;
    f.h[0] = *(const v8bf*)(p);
    f.h[1] = *(const v8bf*)(p + 16);
    return f.v;
}

// ---------------- f32 -> bf16 weight conversion ----------------
__global__ void cvt_kernel(const float* __restrict__ src, __bf16* __restrict__ dst, int n) {
    int i = blockIdx.x * 256 + threadIdx.x;
    if (i < n) dst[i] = f2bf(src[i]);
}

// ---------------- w_out pad (24->32 rows) + bias sums ----------------
__global__ void prep_kernel(const float* __restrict__ w_out,
                            const float* __restrict__ bihf, const float* __restrict__ bhhf,
                            const float* __restrict__ bihb, const float* __restrict__ bhhb,
                            __bf16* __restrict__ Wo, float* __restrict__ bias_f,
                            float* __restrict__ bias_b) {
    int i = blockIdx.x * 256 + threadIdx.x;
    if (i < KP_ * 512) {
        int n = i >> 9, kk = i & 511;
        Wo[i] = (n < K_) ? f2bf(w_out[n * 512 + kk]) : f2bf(0.0f);
    }
    if (i < G4_)            bias_f[i] = bihf[i] + bhhf[i];
    else if (i < 2 * G4_)   bias_b[i - G4_] = bihb[i - G4_] + bhhb[i - G4_];
}

// ---------------- embedding gather -> bf16 X[t*B+b][E] ----------------
__global__ void embed_kernel(const int* __restrict__ sent, const float* __restrict__ emb,
                             __bf16* __restrict__ Xbf) {
    int r = blockIdx.x;             // r = t*B + b
    int t = r >> 5, b = r & 31;
    int tok = sent[b * T_ + t];     // sentence is [B,T]
    Xbf[(size_t)r * E_ + threadIdx.x] = f2bf(emb[(size_t)tok * E_ + threadIdx.x]);
}

// ---------------- G = X @ W_ih^T + (b_ih+b_hh), both directions ----------------
// grid (1024 M-tiles, 16); block 256 = 8 waves; wave -> one 16x16 tile, K=256.
__global__ void __launch_bounds__(256)
gates_gemm_kernel(const __bf16* __restrict__ Xbf,
                  const __bf16* __restrict__ Wf, const __bf16* __restrict__ Wb,
                  const float* __restrict__ bias_f, const float* __restrict__ bias_b,
                  float* __restrict__ Gf, float* __restrict__ Gb) {
    int lane = threadIdx.x & 31;
    int wave = threadIdx.x >> 5;
    int ntg  = blockIdx.y * 8 + wave;          // 0..127 (64 N-tiles x 2 dirs)
    int dir  = ntg >> 6;
    int nb   = (ntg & 63) * 16;
    const __bf16* W    = dir ? Wb : Wf;
    const float*  bias = dir ? bias_b : bias_f;
    float*        G    = dir ? Gb : Gf;
    int r0    = blockIdx.x * 16;
    int half8 = (lane >> 4) * 8;
    int l15   = lane & 15;
    const __bf16* Ab = Xbf + (size_t)(r0 + l15) * E_;
    const __bf16* Bb = W   + (size_t)(nb + l15) * E_;
    v8f acc = {};
#pragma unroll
    for (int kb = 0; kb < 8; ++kb) {
        int k = kb * 32 + half8;
        v16bf a = load_frag(Ab + k);
        v16bf b = load_frag(Bb + k);
        acc = __builtin_amdgcn_wmma_f32_16x16x32_bf16(false, a, false, b, (short)0,
                                                      acc, false, false);
    }
    int col = nb + l15;
    float bc = bias[col];
#pragma unroll
    for (int vr = 0; vr < 8; ++vr) {
        int row = r0 + vr + half8;
        G[(size_t)row * G4_ + col] = acc[vr] + bc;
    }
}

// ---------------- recurrent LSTM, one 1024-thread workgroup per direction ----------
// 32 waves: wave = (wg<<1)|m; M-tile m covers batch rows m*16..+15; column group
// wg in [0,16) covers hidden cols [wg*16, wg*16+16) for ALL FOUR gates, so the
// LSTM elementwise (i/f/g/o, c-state) stays entirely in this wave's registers.
// Only the 16KB bf16 h-state round-trips through LDS each step.
__global__ void __launch_bounds__(1024)
lstm_rec_kernel(const float* __restrict__ Gf, const float* __restrict__ Gb,
                const __bf16* __restrict__ Whf, const __bf16* __restrict__ Whb,
                const float* __restrict__ h0, const float* __restrict__ c0,
                __bf16* __restrict__ Hf, __bf16* __restrict__ Hb) {
    __shared__ __attribute__((aligned(16))) __bf16 hlds[B_][H_ + 8];
    int dir = blockIdx.x;
    const float*  G    = dir ? Gb : Gf;
    const __bf16* Wh   = dir ? Whb : Whf;
    __bf16*       Hout = dir ? Hb : Hf;
    int tid = threadIdx.x;
    int lane = tid & 31, wave = tid >> 5;      // 32 waves
    int m = wave & 1, wg = wave >> 1;          // wg in [0,16)
    int half8 = (lane >> 4) * 8;
    int l15   = lane & 15;
    int col   = wg * 16 + l15;                 // this lane's hidden column

    // init h (LDS, bf16) and c (registers)
    for (int i = tid; i < B_ * H_; i += 1024) {
        int b = i >> 8, j = i & 255;
        hlds[b][j] = f2bf(h0[dir * (B_ * H_) + i]);
    }
    float cst[8];
#pragma unroll
    for (int vr = 0; vr < 8; ++vr) {
        int row = m * 16 + vr + half8;
        cst[vr] = c0[dir * (B_ * H_) + row * H_ + col];
    }
    __syncthreads();

    for (int step = 0; step < T_; ++step) {
        int t = dir ? (T_ - 1 - step) : step;
        const float* Gt = G + (size_t)t * (B_ * G4_);
        v8f acc[4];
#pragma unroll
        for (int g = 0; g < 4; ++g) {
            int gcol = g * 256 + col;
#pragma unroll
            for (int vr = 0; vr < 8; ++vr)
                acc[g][vr] = Gt[(m * 16 + vr + half8) * G4_ + gcol];
        }
#pragma unroll
        for (int kb = 0; kb < 8; ++kb) {
            int k = kb * 32 + half8;
            v16bf a = load_frag(&hlds[m * 16 + l15][k]);
#pragma unroll
            for (int g = 0; g < 4; ++g) {
                int n = g * 256 + wg * 16 + l15;
                v16bf b = load_frag(Wh + (size_t)n * H_ + k);
                acc[g] = __builtin_amdgcn_wmma_f32_16x16x32_bf16(
                    false, a, false, b, (short)0, acc[g], false, false);
            }
        }
        float hval[8];
#pragma unroll
        for (int vr = 0; vr < 8; ++vr) {
            float iv = sigm(acc[0][vr]);
            float fv = sigm(acc[1][vr]);
            float gv = tanh_fast(acc[2][vr]);
            float ov = sigm(acc[3][vr]);
            float c  = fv * cst[vr] + iv * gv;
            cst[vr]  = c;
            hval[vr] = ov * tanh_fast(c);
        }
        __syncthreads();   // all reads of old h done before overwriting
        __bf16* Ht = Hout + (size_t)t * (B_ * H_);
#pragma unroll
        for (int vr = 0; vr < 8; ++vr) {
            int row = m * 16 + vr + half8;
            __bf16 hb = f2bf(hval[vr]);
            hlds[row][col] = hb;
            Ht[row * H_ + col] = hb;
        }
        __syncthreads();   // new h visible before next step's fragment loads
    }
}

// ---------------- feats = concat(Hf,Hb) @ Wo^T + b_out ----------------
__global__ void __launch_bounds__(64)
feats_gemm_kernel(const __bf16* __restrict__ Hf, const __bf16* __restrict__ Hb,
                  const __bf16* __restrict__ Wo, const float* __restrict__ b_out,
                  float* __restrict__ feats) {
    int lane = threadIdx.x & 31;
    int wave = threadIdx.x >> 5;        // 0..1 -> N-tile
    int nb = wave * 16;
    int r0 = blockIdx.x * 16;
    int half8 = (lane >> 4) * 8;
    int l15   = lane & 15;
    int arow  = r0 + l15;
    const __bf16* Bb = Wo + (size_t)(nb + l15) * 512;
    v8f acc = {};
#pragma unroll
    for (int ks = 0; ks < 16; ++ks) {
        int k = ks * 32 + half8;
        const __bf16* Asrc = (ks < 8) ? (Hf + (size_t)arow * H_ + k)
                                      : (Hb + (size_t)arow * H_ + (k - 256));
        v16bf a = load_frag(Asrc);
        v16bf b = load_frag(Bb + k);
        acc = __builtin_amdgcn_wmma_f32_16x16x32_bf16(false, a, false, b, (short)0,
                                                      acc, false, false);
    }
    int col = nb + l15;
    if (col < K_) {
        float bo = b_out[col];
#pragma unroll
        for (int vr = 0; vr < 8; ++vr)
            feats[(size_t)(r0 + vr + half8) * K_ + col] = acc[vr] + bo;
    }
}

// ---------------- Viterbi: one block, threads (b=32, next=24) ----------------
__global__ void __launch_bounds__(768)
viterbi_kernel(const float* __restrict__ feats, const float* __restrict__ trans,
               unsigned char* __restrict__ bp, float* __restrict__ out) {
    __shared__ float tl[K_][K_];
    __shared__ float fv[2][B_][K_];
    int b = threadIdx.x;
    int k = threadIdx.y;
    int tid = threadIdx.y * 32 + threadIdx.x;
    for (int i = tid; i < K_ * K_; i += 768) tl[i / K_][i % K_] = trans[i];
    fv[0][b][k] = (k == START_) ? 0.0f : NEGV;
    __syncthreads();
    int p = 0;
    for (int t = 0; t < T_; ++t) {
        float best = -3.4e38f; int arg = 0;
#pragma unroll 4
        for (int prev = 0; prev < K_; ++prev) {
            float s = fv[p][b][prev] + tl[k][prev];
            if (s > best) { best = s; arg = prev; }   // first max wins (argmax semantics)
        }
        fv[1 - p][b][k] = best + feats[(size_t)(t * B_ + b) * K_ + k];
        bp[(size_t)(t * B_ + b) * K_ + k] = (unsigned char)arg;
        __syncthreads();
        p ^= 1;
    }
    __threadfence_block();   // order block's global bp writes before backtrack reads
    __syncthreads();
    if (k == 0) {
        float best = -3.4e38f; int arg = 0;
        for (int kk = 0; kk < K_; ++kk) {
            float s = fv[p][b][kk] + tl[STOP_][kk];
            if (s > best) { best = s; arg = kk; }
        }
        out[b] = best;                         // path_score
        int tag = arg;
        for (int t = T_ - 1; t >= 0; --t) {
            out[B_ + b * T_ + t] = (float)tag;  // path[b][t]
            tag = bp[(size_t)(t * B_ + b) * K_ + tag];
        }
    }
}

extern "C" void kernel_launch(void* const* d_in, const int* in_sizes, int n_in,
                              void* d_out, int out_size, void* d_ws, size_t ws_size,
                              hipStream_t stream) {
    (void)in_sizes; (void)n_in; (void)out_size; (void)ws_size;
    const int*   sent = (const int*)d_in[0];
    const float* emb  = (const float*)d_in[1];
    const float* wihf = (const float*)d_in[2];
    const float* whhf = (const float*)d_in[3];
    const float* bihf = (const float*)d_in[4];
    const float* bhhf = (const float*)d_in[5];
    const float* wihb = (const float*)d_in[6];
    const float* whhb = (const float*)d_in[7];
    const float* bihb = (const float*)d_in[8];
    const float* bhhb = (const float*)d_in[9];
    const float* wout = (const float*)d_in[10];
    const float* bout = (const float*)d_in[11];
    const float* trans = (const float*)d_in[12];
    const float* h0   = (const float*)d_in[13];
    const float* c0   = (const float*)d_in[14];
    float* out = (float*)d_out;

    char* ws = (char*)d_ws;
    size_t off = 0;
    auto alloc = [&](size_t bytes) -> char* {
        off = (off + 255) & ~(size_t)255;
        char* p = ws + off; off += bytes; return p;
    };
    __bf16* Xbf  = (__bf16*)alloc((size_t)T_ * B_ * E_ * 2);
    __bf16* Wif  = (__bf16*)alloc((size_t)G4_ * E_ * 2);
    __bf16* Whf  = (__bf16*)alloc((size_t)G4_ * H_ * 2);
    __bf16* Wib  = (__bf16*)alloc((size_t)G4_ * E_ * 2);
    __bf16* Whb  = (__bf16*)alloc((size_t)G4_ * H_ * 2);
    __bf16* Wo   = (__bf16*)alloc((size_t)KP_ * 512 * 2);
    float*  biasf = (float*)alloc((size_t)G4_ * 4);
    float*  biasb = (float*)alloc((size_t)G4_ * 4);
    float*  Gfp  = (float*)alloc((size_t)T_ * B_ * G4_ * 4);
    float*  Gbp  = (float*)alloc((size_t)T_ * B_ * G4_ * 4);
    __bf16* Hf   = (__bf16*)alloc((size_t)T_ * B_ * H_ * 2);
    __bf16* Hb   = (__bf16*)alloc((size_t)T_ * B_ * H_ * 2);
    float*  feats = (float*)alloc((size_t)T_ * B_ * K_ * 4);
    unsigned char* bp = (unsigned char*)alloc((size_t)T_ * B_ * K_);

    int nW = G4_ * E_;
    cvt_kernel<<<(nW + 255) / 256, 256, 0, stream>>>(wihf, Wif, nW);
    cvt_kernel<<<(nW + 255) / 256, 256, 0, stream>>>(whhf, Whf, nW);
    cvt_kernel<<<(nW + 255) / 256, 256, 0, stream>>>(wihb, Wib, nW);
    cvt_kernel<<<(nW + 255) / 256, 256, 0, stream>>>(whhb, Whb, nW);
    prep_kernel<<<64, 256, 0, stream>>>(wout, bihf, bhhf, bihb, bhhb, Wo, biasf, biasb);
    embed_kernel<<<T_ * B_, 256, 0, stream>>>(sent, emb, Xbf);
    gates_gemm_kernel<<<dim3(1024, 16), 256, 0, stream>>>(Xbf, Wif, Wib, biasf, biasb,
                                                          Gfp, Gbp);
    lstm_rec_kernel<<<2, 1024, 0, stream>>>(Gfp, Gbp, Whf, Whb, h0, c0, Hf, Hb);
    feats_gemm_kernel<<<1024, 64, 0, stream>>>(Hf, Hb, Wo, bout, feats);
    viterbi_kernel<<<1, dim3(32, K_), 0, stream>>>(feats, trans, bp, out);
}
